// Pc_52682068853450
// MI455X (gfx1250) — matrix operand order repaired and checked by reading, not compiled, not run on hardware
//
#include <hip/hip_runtime.h>
#include <hip/hip_bf16.h>
#include <math.h>

typedef __bf16 bf16;
typedef __attribute__((ext_vector_type(8)))  __bf16 v8bf;
typedef __attribute__((ext_vector_type(16))) __bf16 v16bf;
typedef __attribute__((ext_vector_type(8)))  float  v8f;

static constexpr int Bn = 256, Cc = 10, Hh = 28, Ww = 28, Ff = 80, Rr = 2;
static constexpr int P  = Bn * Hh * Ww;          // 200704 pixels
static constexpr int NT = P / 16;                // 12544 N-tiles (784 % 16 == 0)
static constexpr float LOG2PI = 1.8378770664093453f;

__device__ __forceinline__ float eluf(float x) {
    return x > 0.f ? x : (expf(x) - 1.f);
}

// ---------------------------------------------------------------------------
// Implicit-GEMM conv via V_WMMA_F32_16X16X32_BF16.
//   KS: spatial kernel size (7/3/1), CT: Cin/32 K-subtiles per tap,
//   MT: number of 16-row M tiles, IC: mask include-center, EPI: epilogue.
// Weights W: [MT*16][KS*KS*CT*32] bf16 row-major (pre-masked, K-ordered).
// Activations act: NHWC bf16, CinS = CT*32 channels per pixel.
// One wave = one 16-pixel N tile, all MT M-tiles (B-fragment reuse).
// ---------------------------------------------------------------------------
template<int KS, int CT, int MT, int IC, int EPI>
__global__ __launch_bounds__(256)
void conv_wmma(const bf16* __restrict__ W,
               const bf16* __restrict__ act,
               const float* __restrict__ bias,
               float* __restrict__ outF, int coutF, int coutValid,
               bf16* __restrict__ outB,
               float* __restrict__ hres)
{
    constexpr int PAD  = KS / 2;
    constexpr int CinS = CT * 32;
    constexpr int Ktot = KS * KS * CinS;

    const int lane  = threadIdx.x & 31;
    const int laneN = lane & 15;           // column (pixel) within tile
    const int hi    = lane >> 4;           // K half selector
    const int c0    = hi * 8;              // C-D row base

    const int ntile = blockIdx.x * 8 + (threadIdx.x >> 5);
    const int pix   = ntile * 16 + laneN;
    const int b     = pix / 784;
    const int pp    = pix - b * 784;
    const int py    = pp / 28;
    const int px    = pp - py * 28;

    const v8f vzero = {0.f,0.f,0.f,0.f,0.f,0.f,0.f,0.f};
    v8f acc[MT];
#pragma unroll
    for (int m = 0; m < MT; ++m) acc[m] = vzero;

    const bf16 zb = (bf16)0.f;
    const v16bf z16 = {zb,zb,zb,zb,zb,zb,zb,zb,zb,zb,zb,zb,zb,zb,zb,zb};

#pragma unroll 1
    for (int patch = 0; patch < KS * KS; ++patch) {
        const int ky = patch / KS, kx = patch - ky * KS;
        // PixelCNN causal mask: taps fully zeroed in the weights are skipped
        // (uniform scalar branch across the wave).
        if (!((ky < PAD) || (ky == PAD && kx < PAD + IC))) continue;
        const int iy = py + ky - PAD;
        const int ix = px + kx - PAD;
        const bool vld = ((unsigned)iy < 28u) && ((unsigned)ix < 28u);
        const bf16* bbase = act + (size_t)(b * 784 + iy * 28 + ix) * CinS;
#pragma unroll 1
        for (int cs = 0; cs < CT; ++cs) {
            // B fragment: 32x16 K-major; lane holds 16 contiguous K values
            // (lanes 0-15 -> K 0..15, lanes 16-31 -> K 16..31). 32B aligned.
            v16bf bfrag = z16;
            if (vld)
                bfrag = *(const v16bf*)(bbase + cs * 32 + hi * 16);

            const int kt = (patch * CT + cs) * 32;
#pragma unroll
            for (int m = 0; m < MT; ++m) {
                // A fragment: row M = laneN of this M-tile, two 8-elem chunks
                // at K offsets {hi*16, hi*16+8} within the 32-wide K tile.
                const bf16* wp = W + (size_t)(m * 16 + laneN) * Ktot + kt + hi * 16;
                const v8bf a0 = *(const v8bf*)(wp);
                const v8bf a1 = *(const v8bf*)(wp + 8);
                const v16bf afrag = __builtin_shufflevector(
                    a0, a1, 0,1,2,3,4,5,6,7,8,9,10,11,12,13,14,15);
                acc[m] = __builtin_amdgcn_wmma_f32_16x16x32_bf16(
                    false, afrag, false, bfrag, (short)0, acc[m], false, false);
            }
        }
    }

    // Epilogues: C/D layout -> lane column = laneN, VGPR v -> row c0+v.
    if (EPI == 0) {                       // bias + f32 store (stem / head out2)
#pragma unroll
        for (int m = 0; m < MT; ++m)
#pragma unroll
            for (int v = 0; v < 8; ++v) {
                const int ch = m * 16 + c0 + v;
                if (ch < coutValid)
                    outF[(size_t)pix * coutF + ch] = acc[m][v] + bias[ch];
            }
    } else if (EPI == 1) {                // bias + concat-ELU -> bf16 NHWC 2*Cout
        constexpr int CO = MT * 16;
#pragma unroll
        for (int m = 0; m < MT; ++m)
#pragma unroll
            for (int v = 0; v < 8; ++v) {
                const int ch = m * 16 + c0 + v;
                const float t = acc[m][v] + bias[ch];
                outB[(size_t)pix * (2 * CO) + ch]      = (bf16)eluf(t);
                outB[(size_t)pix * (2 * CO) + CO + ch] = (bf16)eluf(-t);
            }
    } else {                              // gated residual: h += a * sigmoid(g)
#pragma unroll
        for (int m = 0; m < MT / 2; ++m)
#pragma unroll
            for (int v = 0; v < 8; ++v) {
                const int ca = m * 16 + c0 + v;
                const float a = acc[m][v]          + bias[ca];
                const float g = acc[m + MT / 2][v] + bias[ca + 80];
                const float s = 1.f / (1.f + expf(-g));
                hres[(size_t)pix * 80 + ca] += a * s;
            }
    }
}

// ---- x (NCHW f32) -> padded NHWC bf16, Cpad = 32 ---------------------------
__global__ __launch_bounds__(256)
void prep_x(const float* __restrict__ x, bf16* __restrict__ xp)
{
    const int p = blockIdx.x * 256 + threadIdx.x;
    if (p >= P) return;
    const int b = p / 784, pp = p - b * 784;
#pragma unroll
    for (int c = 0; c < 32; ++c)
        xp[(size_t)p * 32 + c] =
            (c < Cc) ? (bf16)x[(size_t)(b * Cc + c) * 784 + pp] : (bf16)0.f;
}

// ---- OIHW f32 weights -> masked, K-ordered bf16 [Mpad][KS*KS*CinPad] -------
__global__ __launch_bounds__(256)
void prep_w(const float* __restrict__ w, bf16* __restrict__ dst,
            int Cout, int Cin, int KS, int CinPad, int Mpad, int inc)
{
    const int Ktot = KS * KS * CinPad;
    const int n = Mpad * Ktot;
    const int i = blockIdx.x * 256 + threadIdx.x;
    if (i >= n) return;
    const int m = i / Ktot, k = i - m * Ktot;
    const int patch = k / CinPad, cin = k - patch * CinPad;
    const int ky = patch / KS, kx = patch - ky * KS;
    const int pad = KS / 2;
    float v = 0.f;
    const bool vis = (ky < pad) || (ky == pad && kx < pad + inc);
    if (m < Cout && cin < Cin && vis)
        v = w[((size_t)(m * Cin + cin) * KS + ky) * KS + kx];
    dst[i] = (bf16)v;
}

// ---- concat-ELU of residual stream h (f32, 80ch) -> bf16 160ch -------------
__global__ __launch_bounds__(256)
void celu_h(const float* __restrict__ h, bf16* __restrict__ a)
{
    const int i = blockIdx.x * 256 + threadIdx.x;
    if (i >= P * Ff) return;
    const int p = i / Ff, c = i - p * Ff;
    const float t = h[(size_t)p * Ff + c];
    a[(size_t)p * (2 * Ff) + c]      = (bf16)eluf(t);
    a[(size_t)p * (2 * Ff) + Ff + c] = (bf16)eluf(-t);
}

// ---- output[0..N) = x ------------------------------------------------------
__global__ __launch_bounds__(256)
void copy_x(const float* __restrict__ x, float* __restrict__ o, int n)
{
    const int i = blockIdx.x * 256 + threadIdx.x;
    if (i < n) o[i] = x[i];
}

// ---- Gaussian NLL, stage 1: per-block partial sums -------------------------
__global__ __launch_bounds__(256)
void nll_part(const float* __restrict__ x, const float* __restrict__ o2,
              float* __restrict__ part)
{
    __shared__ float s[256];
    const int p = blockIdx.x * 256 + threadIdx.x;   // exactly P threads total
    const int b = p / 784, pp = p - b * 784;
    float sum = 0.f;
#pragma unroll
    for (int c = 0; c < Cc; ++c) {
        const float mean = o2[(size_t)p * 20 + c];
        const float ls   = o2[(size_t)p * 20 + 10 + c];
        const float xv   = x[(size_t)(b * Cc + c) * 784 + pp];
        const float z    = (xv - mean) * expf(-ls);
        sum += 0.5f * z * z + ls + 0.5f * LOG2PI;
    }
    s[threadIdx.x] = sum;
    __syncthreads();
    for (int st = 128; st > 0; st >>= 1) {
        if (threadIdx.x < st) s[threadIdx.x] += s[threadIdx.x + st];
        __syncthreads();
    }
    if (threadIdx.x == 0) part[blockIdx.x] = s[0];
}

// ---- stage 2: deterministic final reduction, write -nll --------------------
__global__ __launch_bounds__(256)
void nll_final(const float* __restrict__ part, int n, float* __restrict__ out)
{
    __shared__ float s[256];
    float sum = 0.f;
    for (int i = threadIdx.x; i < n; i += 256) sum += part[i];
    s[threadIdx.x] = sum;
    __syncthreads();
    for (int st = 128; st > 0; st >>= 1) {
        if (threadIdx.x < st) s[threadIdx.x] += s[threadIdx.x + st];
        __syncthreads();
    }
    if (threadIdx.x == 0) *out = -s[0];
}

// ---------------------------------------------------------------------------
extern "C" void kernel_launch(void* const* d_in, const int* in_sizes, int n_in,
                              void* d_out, int out_size, void* d_ws, size_t ws_size,
                              hipStream_t stream)
{
    const float* x      = (const float*)d_in[0];
    const float* w_in   = (const float*)d_in[1];
    const float* b_in   = (const float*)d_in[2];
    const float* blk_w1 = (const float*)d_in[3];
    const float* blk_b1 = (const float*)d_in[4];
    const float* blk_w2 = (const float*)d_in[5];
    const float* blk_b2 = (const float*)d_in[6];
    const float* w_out1 = (const float*)d_in[7];
    const float* b_out1 = (const float*)d_in[8];
    const float* w_out2 = (const float*)d_in[9];
    const float* b_out2 = (const float*)d_in[10];
    float* out = (float*)d_out;

    // Workspace carve-up (256B aligned).
    char* ws = (char*)d_ws;
    size_t off = 0;
    auto carve = [&](size_t bytes) {
        void* p = ws + off;
        off = (off + bytes + 255) & ~(size_t)255;
        return p;
    };
    bf16*  xpad  = (bf16*)carve((size_t)P * 32 * 2);            // stem input, Cpad=32
    float* hbuf  = (float*)carve((size_t)P * Ff * 4);           // residual stream (f32)
    bf16*  actA  = (bf16*)carve((size_t)P * 160 * 2);           // concat-ELU buffers
    bf16*  actB  = (bf16*)carve((size_t)P * 160 * 2);
    bf16*  wstem = (bf16*)carve((size_t)80 * 1568 * 2);         // 7x7, K = 49*32
    bf16*  wblk1 = (bf16*)carve((size_t)Rr * 80 * 1440 * 2);    // 3x3, K = 9*160
    bf16*  wblk2 = (bf16*)carve((size_t)Rr * 160 * 1440 * 2);
    bf16*  wo1   = (bf16*)carve((size_t)80 * 160 * 2);          // 1x1
    bf16*  wo2   = (bf16*)carve((size_t)32 * 160 * 2);          // 1x1, Mpad 20->32
    float* part  = (float*)carve((size_t)(P / 256) * 4);
    float* o2    = hbuf;  // overlay: h is dead after the last concat-ELU

    const int NX = Bn * Cc * Hh * Ww;                           // 2,007,040
    copy_x<<<(NX + 255) / 256, 256, 0, stream>>>(x, out, NX);

    prep_x<<<P / 256, 256, 0, stream>>>(x, xpad);

    prep_w<<<(80 * 1568 + 255) / 256, 256, 0, stream>>>(w_in, wstem, 80, 10, 7, 32, 80, 0);
    for (int r = 0; r < Rr; ++r) {
        prep_w<<<(80 * 1440 + 255) / 256, 256, 0, stream>>>(
            blk_w1 + (size_t)r * 80 * 160 * 9, wblk1 + (size_t)r * 80 * 1440,
            80, 160, 3, 160, 80, 1);
        prep_w<<<(160 * 1440 + 255) / 256, 256, 0, stream>>>(
            blk_w2 + (size_t)r * 160 * 160 * 9, wblk2 + (size_t)r * 160 * 1440,
            160, 160, 3, 160, 160, 1);
    }
    prep_w<<<(80 * 160 + 255) / 256, 256, 0, stream>>>(w_out1, wo1, 80, 160, 1, 160, 80, 1);
    prep_w<<<(32 * 160 + 255) / 256, 256, 0, stream>>>(w_out2, wo2, 20, 160, 1, 160, 32, 1);

    const int cgrid = NT / 8;                                   // 8 waves/block
    // Stem: masked 7x7, 10(pad32) -> 80, f32 store into residual stream h.
    conv_wmma<7, 1, 5, 0, 0><<<cgrid, 256, 0, stream>>>(
        wstem, xpad, b_in, hbuf, 80, 80, nullptr, nullptr);

    for (int r = 0; r < Rr; ++r) {
        celu_h<<<(P * Ff + 255) / 256, 256, 0, stream>>>(hbuf, actA);
        // conv1: 160 -> 80, fused concat-ELU -> actB (bf16, 160ch)
        conv_wmma<3, 5, 5, 1, 1><<<cgrid, 256, 0, stream>>>(
            wblk1 + (size_t)r * 80 * 1440, actA, blk_b1 + (size_t)r * 80,
            nullptr, 0, 0, actB, nullptr);
        // conv2: 160 -> 160, fused gate: h += a * sigmoid(g)
        conv_wmma<3, 5, 10, 1, 2><<<cgrid, 256, 0, stream>>>(
            wblk2 + (size_t)r * 160 * 1440, actB, blk_b2 + (size_t)r * 160,
            nullptr, 0, 0, nullptr, hbuf);
    }

    celu_h<<<(P * Ff + 255) / 256, 256, 0, stream>>>(hbuf, actA);
    // out1: 1x1, 160 -> 80, fused concat-ELU -> actB
    conv_wmma<1, 5, 5, 1, 1><<<cgrid, 256, 0, stream>>>(
        wo1, actA, b_out1, nullptr, 0, 0, actB, nullptr);
    // out2: 1x1, 160 -> 20 (Mpad 32), f32 store -> o2 (overlays h)
    conv_wmma<1, 5, 2, 1, 0><<<cgrid, 256, 0, stream>>>(
        wo2, actB, b_out2, o2, 20, 20, nullptr, nullptr);

    nll_part<<<P / 256, 256, 0, stream>>>(x, o2, part);
    nll_final<<<1, 256, 0, stream>>>(part, P / 256, out + (out_size - 1));
}
